// BitLinear_11579231830568
// MI455X (gfx1250) — compile-verified, matrix operand order checked by reading, not run on hardware
//
#include <hip/hip_runtime.h>

typedef unsigned short ushort_t;
typedef unsigned int uint32;

typedef __attribute__((ext_vector_type(16))) __bf16 v16bf;
typedef __attribute__((ext_vector_type(8)))  float  v8f;

union Frag { uint4 q[2]; v16bf v; };

#if defined(__has_builtin)
#if __has_builtin(__builtin_amdgcn_global_load_async_to_lds_b128)
#define HAVE_ASYNC_LDS 1
#endif
#endif
#ifndef HAVE_ASYNC_LDS
#define HAVE_ASYNC_LDS 0
#endif

#if HAVE_ASYNC_LDS
// Param types per clang diagnostic: 16B int vector, AS1 (global) source,
// AS3 (LDS) destination.
typedef int v4i __attribute__((__vector_size__(16)));
typedef __attribute__((address_space(1))) v4i gbl_v4i;
typedef __attribute__((address_space(3))) v4i lds_v4i;

__device__ __forceinline__ gbl_v4i* to_gbl(const ushort_t* p) {
  return (gbl_v4i*)(unsigned long long)p;
}
__device__ __forceinline__ lds_v4i* to_lds(const ushort_t* p) {
  return (lds_v4i*)(unsigned int)(unsigned long long)p;  // low 32b = LDS offset
}
#endif

__device__ __forceinline__ ushort_t f2bf(float f) {
  union { float f; uint32 u; } c; c.f = f;
  uint32 u = c.u;
  u += 0x7FFFu + ((u >> 16) & 1u);   // round-to-nearest-even
  return (ushort_t)(u >> 16);
}

// ---------------------------------------------------------------------------
// Pass 1: per-row absmean -> ternarize weight into bf16 {-1,0,+1}, emit scale.
// ---------------------------------------------------------------------------
__global__ __launch_bounds__(256) void ternarize_kernel(
    const float* __restrict__ w, ushort_t* __restrict__ wt,
    float* __restrict__ scale)
{
  const int row = blockIdx.x;
  const int tid = threadIdx.x;
  const size_t base = (size_t)row * 4096 + (size_t)tid * 16;

  float v[16];
  const float4* p = (const float4*)(w + base);
  #pragma unroll
  for (int i = 0; i < 4; ++i) {
    float4 f = p[i];
    v[i*4+0] = f.x; v[i*4+1] = f.y; v[i*4+2] = f.z; v[i*4+3] = f.w;
  }
  float s = 0.f;
  #pragma unroll
  for (int i = 0; i < 16; ++i) s += fabsf(v[i]);

  __shared__ float red[256];
  red[tid] = s;
  __syncthreads();
  for (int off = 128; off > 0; off >>= 1) {
    if (tid < off) red[tid] += red[tid + off];
    __syncthreads();
  }
  const float absmean = red[0] * (1.0f / 4096.0f);
  const float thr = 0.7f * absmean;

  ushort_t t[16];
  #pragma unroll
  for (int i = 0; i < 16; ++i) {
    const float a = fabsf(v[i]);
    t[i] = (a > thr) ? (v[i] > 0.f ? (ushort_t)0x3F80u : (ushort_t)0xBF80u)
                     : (ushort_t)0;
  }
  uint32 pk[8];
  #pragma unroll
  for (int i = 0; i < 8; ++i)
    pk[i] = (uint32)t[2*i] | ((uint32)t[2*i+1] << 16);
  uint4* q = (uint4*)(wt + base);
  q[0] = make_uint4(pk[0], pk[1], pk[2], pk[3]);
  q[1] = make_uint4(pk[4], pk[5], pk[6], pk[7]);
  if (tid == 0) scale[row] = absmean;
}

// ---------------------------------------------------------------------------
// Pass 2: X fp32 -> bf16 (128MB panel becomes L2-resident, 192MB L2).
// ---------------------------------------------------------------------------
__global__ __launch_bounds__(256) void cvt_kernel(const float* __restrict__ x,
                                                  ushort_t* __restrict__ xb,
                                                  int n8)
{
  const int i = blockIdx.x * 256 + threadIdx.x;
  if (i >= n8) return;
  const float4* p = (const float4*)x + (size_t)i * 2;
  const float4 a = p[0], b = p[1];
  uint32 pk[4];
  pk[0] = (uint32)f2bf(a.x) | ((uint32)f2bf(a.y) << 16);
  pk[1] = (uint32)f2bf(a.z) | ((uint32)f2bf(a.w) << 16);
  pk[2] = (uint32)f2bf(b.x) | ((uint32)f2bf(b.y) << 16);
  pk[3] = (uint32)f2bf(b.z) | ((uint32)f2bf(b.w) << 16);
  ((uint4*)xb)[i] = make_uint4(pk[0], pk[1], pk[2], pk[3]);
}

// ---------------------------------------------------------------------------
// Pass 3: WMMA GEMM. 256x128 block tile, 8 wave32s, 64x64 wave tile (4x4
// accumulators -> 16 wmma per K-step per wave). Double-buffered LDS with
// async global->LDS DMA (ASYNCcnt); one barrier per K-step; K-loop kept
// rolled so VGPR tuples stay below 256 (no MSB windows / acc copies).
// ---------------------------------------------------------------------------
#define BM 256
#define BN 128
#define BK 32
#define LDT 40   // 80B row stride: 16-lane b128 frag reads hit all 64 banks

__global__ __launch_bounds__(256) void bitlinear_gemm(
    const ushort_t* __restrict__ xb, const ushort_t* __restrict__ wt,
    const float* __restrict__ scale, float* __restrict__ out)
{
  __shared__ __align__(16) ushort_t As[2][BM * LDT];
  __shared__ __align__(16) ushort_t Bs[2][BN * LDT];

  const int tid  = threadIdx.x;
  const int m0   = blockIdx.y * BM;
  const int n0   = blockIdx.x * BN;
  const int wave = tid >> 5;
  const int lane = tid & 31;
  const int hf   = lane >> 4;         // half-wave (ISA A/B frag layout)
  const int r    = lane & 15;
  const int wm   = (wave & 3) * 64;   // 4 waves along M
  const int wn   = (wave >> 2) * 64;  // 2 waves along N

  v8f acc[4][4] = {};   // [j][i] : j = N tile, i = M tile

  // staging geometry: A row t -> 4x b128 ; B row t>>1 half (t&1) -> 2x b128
  const int arow = tid;
  const int brow = tid >> 1;
  const int bseg = (tid & 1) * 16;    // ushort offset
  const ushort_t* gA = xb + (size_t)(m0 + arow) * 4096;
  const ushort_t* gB = wt + (size_t)(n0 + brow) * 4096 + bseg;

  auto stage = [&](int ks, int buf) {
    const int k0 = ks * BK;
#if HAVE_ASYNC_LDS
    gbl_v4i* ga = to_gbl(gA + k0);
    gbl_v4i* gb = to_gbl(gB + k0);
    lds_v4i* la = to_lds(&As[buf][arow * LDT]);
    lds_v4i* lb = to_lds(&Bs[buf][brow * LDT + bseg]);
    __builtin_amdgcn_global_load_async_to_lds_b128(ga, la,  0, 0);
    __builtin_amdgcn_global_load_async_to_lds_b128(ga, la, 16, 0);
    __builtin_amdgcn_global_load_async_to_lds_b128(ga, la, 32, 0);
    __builtin_amdgcn_global_load_async_to_lds_b128(ga, la, 48, 0);
    __builtin_amdgcn_global_load_async_to_lds_b128(gb, lb,  0, 0);
    __builtin_amdgcn_global_load_async_to_lds_b128(gb, lb, 16, 0);
#else
    uint4 ra[4], rb[2];
    const uint4* pa = (const uint4*)(gA + k0);
    const uint4* pb = (const uint4*)(gB + k0);
    #pragma unroll
    for (int c = 0; c < 4; ++c) ra[c] = pa[c];
    #pragma unroll
    for (int c = 0; c < 2; ++c) rb[c] = pb[c];
    uint4* la = (uint4*)&As[buf][arow * LDT];
    uint4* lb = (uint4*)&Bs[buf][brow * LDT + bseg];
    #pragma unroll
    for (int c = 0; c < 4; ++c) la[c] = ra[c];
    #pragma unroll
    for (int c = 0; c < 2; ++c) lb[c] = rb[c];
#endif
  };

  const int nK = 4096 / BK;   // 128
  stage(0, 0);

  #pragma unroll 1
  for (int ks = 0; ks < nK; ++ks) {
    const int cur = ks & 1;
#if HAVE_ASYNC_LDS
    asm volatile("s_wait_asynccnt 0x0" ::: "memory");  // own DMA visible
#endif
    __syncthreads();            // all waves: buf[cur] filled, buf[cur^1] free
    if (ks + 1 < nK) stage(ks + 1, cur ^ 1);

    // A 16x32 bf16 frag: lanes 0-15 K{0..7,16..23}, lanes 16-31 K{8..15,24..31}
    Frag fa[4];
    #pragma unroll
    for (int i = 0; i < 4; ++i) {
      const ushort_t* rp = &As[cur][(wm + i*16 + r) * LDT + hf * 8];
      fa[i].q[0] = *(const uint4*)rp;
      fa[i].q[1] = *(const uint4*)(rp + 16);
    }
    // stream B fragments one at a time to cap live VGPR tuples
    #pragma unroll
    for (int j = 0; j < 4; ++j) {
      Frag fb;
      const ushort_t* rp = &Bs[cur][(wn + j*16 + r) * LDT + hf * 8];
      fb.q[0] = *(const uint4*)rp;
      fb.q[1] = *(const uint4*)(rp + 16);
      #pragma unroll
      for (int i = 0; i < 4; ++i)
        acc[j][i] = __builtin_amdgcn_wmma_f32_16x16x32_bf16(
            false, fa[i].v, false, fb.v, (short)0, acc[j][i], false, false);
    }
  }

  // C layout: VGPR v -> M = hf*8 + v, N = lane&15 ; apply per-row scale.
  #pragma unroll
  for (int j = 0; j < 4; ++j) {
    const int gn = n0 + wn + j*16 + r;
    const float sc = scale[gn];
    #pragma unroll
    for (int i = 0; i < 4; ++i) {
      float* op = out + (size_t)(m0 + wm + i*16 + hf*8) * 4096 + gn;
      #pragma unroll
      for (int v = 0; v < 8; ++v)
        op[(size_t)v * 4096] = acc[j][i][v] * sc;
    }
  }
}

extern "C" void kernel_launch(void* const* d_in, const int* in_sizes, int n_in,
                              void* d_out, int out_size, void* d_ws, size_t ws_size,
                              hipStream_t stream) {
  const float* x = (const float*)d_in[0];   // [8,2048,4096] fp32
  const float* w = (const float*)d_in[1];   // [4096,4096] fp32
  float* out = (float*)d_out;               // [8,2048,4096] fp32

  char* ws = (char*)d_ws;
  float*    scale = (float*)ws;                                       // 16 KB
  ushort_t* wt    = (ushort_t*)(ws + 16384);                          // 32 MB
  ushort_t* xbf   = (ushort_t*)(ws + 16384 + (size_t)4096*4096*2);    // 128 MB

  ternarize_kernel<<<4096, 256, 0, stream>>>(w, wt, scale);

  const int n8 = (8 * 2048 * 4096) / 8;
  cvt_kernel<<<n8 / 256, 256, 0, stream>>>(x, xbf, n8);

  dim3 grid(4096 / BN, 16384 / BM);         // (32, 64)
  bitlinear_gemm<<<grid, 256, 0, stream>>>(xbf, wt, scale, out);

  (void)in_sizes; (void)n_in; (void)out_size; (void)ws_size;
}
